// RelationAwareAttention_90598040141845
// MI455X (gfx1250) — matrix operand order repaired and checked by reading, not compiled, not run on hardware
//
#include <hip/hip_runtime.h>
#include <hip/hip_bf16.h>

typedef _Float16 f16;
typedef __attribute__((ext_vector_type(16))) _Float16 v16h;
typedef __attribute__((ext_vector_type(8)))  float    v8f;
typedef __attribute__((ext_vector_type(4)))  int      i4v;

#define AS1 __attribute__((address_space(1)))
#define AS3 __attribute__((address_space(3)))

#define B_  2
#define N_  2048
#define D_  1024
#define H_  16
#define HD_ 64

union F16Frag { uint4 u[2]; v16h v; };
union F16x8   { uint4 u;    f16 h[8]; };

static __device__ __forceinline__ v8f wmma_f16(v16h a, v16h b, v8f c) {
  // D = A(16x32 f16) * B(32x16 f16) + C(16x16 f32)
  return __builtin_amdgcn_wmma_f32_16x16x32_f16(false, a, false, b, (short)0, c,
                                                false, false);
}

// ---------------------------------------------------------------- f32 -> f16
__global__ void cvt_f32_f16(const float* __restrict__ in, f16* __restrict__ out,
                            int n) {
  int i = blockIdx.x * blockDim.x + threadIdx.x;
  if (i < n) out[i] = (f16)in[i];
}

// ------------------------------------------------- QKV projection GEMM (WMMA)
// out[m,n] = sum_k x[m,k] * W[n,k] + b[n]; M = B*N = 4096, N = K = D = 1024.
// Each wave computes a 16x64 strip (4 WMMA tiles sharing one A-fragment);
// the 64-wide strip spans exactly one head. grid.z selects Q / K / V.
// All fragment loads are issued before the 4-WMMA chain so the backend can
// clause them and overlap next-iteration loads with the matrix ops.
__global__ __launch_bounds__(256) void gemm_qkv(
    const f16* __restrict__ xh,
    const f16* __restrict__ wqh, const f16* __restrict__ wkh,
    const f16* __restrict__ wvh,
    const float* __restrict__ bq, const float* __restrict__ bk,
    const float* __restrict__ bv,
    f16* __restrict__ qh, f16* __restrict__ kh, f16* __restrict__ vh) {
  const int which = blockIdx.z;
  const f16*   W    = (which == 0) ? wqh : (which == 1) ? wkh : wvh;
  const float* bias = (which == 0) ? bq  : (which == 1) ? bk  : bv;
  f16*         dst  = (which == 0) ? qh  : (which == 1) ? kh  : vh;

  const int lane = threadIdx.x & 31;
  const int wave = threadIdx.x >> 5;
  const int gw   = blockIdx.x * 8 + wave;       // 4096 strips total
  const int mt   = gw >> 4;                     // 0..255  (M tile)
  const int ns   = gw & 15;                     // 0..15   (64-wide N strip)
  const int ln   = lane & 15;
  const int hi   = lane >> 4;                   // 0: lanes 0-15, 1: lanes 16-31
  const int sA   = hi ? 8 : 0;

  const int arow = mt * 16 + ln;                // A row for this lane

  v8f acc[4] = {};
  const f16* ap  = xh + (size_t)arow * D_ + sA;
  const f16* bp0 = W + (size_t)(ns * 64 + ln) * D_ + hi * 8;  // t=0 column
  for (int kk = 0; kk < D_; kk += 32) {
    F16Frag a, b[4];
    a.u[0] = *(const uint4*)(ap + kk);
    a.u[1] = *(const uint4*)(ap + kk + 16);
#pragma unroll
    for (int t = 0; t < 4; ++t) {
      const f16* bp = bp0 + (size_t)t * 16 * D_ + kk;
      b[t].u[0] = *(const uint4*)(bp);
      b[t].u[1] = *(const uint4*)(bp + 16);
    }
#pragma unroll
    for (int t = 0; t < 4; ++t) acc[t] = wmma_f16(a.v, b[t].v, acc[t]);
  }

  const int h = ns;                             // head (strip == head)
#pragma unroll
  for (int t = 0; t < 4; ++t) {
    const int   hd = t * 16 + ln;
    const float bb = bias[ns * 64 + hd];
#pragma unroll
    for (int r = 0; r < 8; ++r) {
      int m   = mt * 16 + r + hi * 8;           // C layout: VGPR r -> row
      int bIx = m >> 11;                        // / N_
      int seq = m & (N_ - 1);
      dst[(((size_t)bIx * H_ + h) * N_ + seq) * HD_ + hd] =
          (f16)(acc[t][r] + bb);
    }
  }
}

// --------------------------------------------- flash attention (WMMA, wave32)
// Block = 8 waves = 128 query rows of one (b,h). Streams K/V in 32-key tiles.
__global__ __launch_bounds__(256) void attn(
    const f16* __restrict__ qh, const f16* __restrict__ kh,
    const f16* __restrict__ vh, const float* __restrict__ pb,
    f16* __restrict__ oh) {
  __shared__ f16 Ks[32][72];       // K tile, row-major  (pad: 144B row stride)
  __shared__ f16 Vt[64][40];       // V tile transposed  (pad: 80B row stride)
  __shared__ f16 Ps[8][16][32];    // per-wave P tile (C-layout -> A-layout)

  const int tid  = threadIdx.x;
  const int lane = tid & 31;
  const int wave = tid >> 5;
  const int ln   = lane & 15;
  const int hi   = lane >> 4;
  const int sA   = hi ? 8 : 0;

  const int bh = blockIdx.y;                    // b*H + h
  const int b  = bh >> 4;
  const int h  = bh & 15;
  const int q0 = blockIdx.x * 128 + wave * 16;  // this wave's query tile

  // Q A-fragments (kept in registers for the whole K loop)
  const f16* qp = qh + ((size_t)bh * N_ + q0 + ln) * HD_;
  F16Frag qf0, qf1;
  qf0.u[0] = *(const uint4*)(qp + sA);
  qf0.u[1] = *(const uint4*)(qp + 16 + sA);
  qf1.u[0] = *(const uint4*)(qp + 32 + sA);
  qf1.u[1] = *(const uint4*)(qp + 48 + sA);

  float mrow[8], lrow[8];
  v8f acc[4] = {};
#pragma unroll
  for (int r = 0; r < 8; ++r) { mrow[r] = -1e30f; lrow[r] = 0.0f; }

  const float scale = 0.125f;                   // 1/sqrt(64)

  // per-thread staging coordinates (32x64 tile, 8 f16 per thread)
  const int krow = tid >> 3;
  const int d0   = (tid & 7) * 8;

  for (int k0 = 0; k0 < N_; k0 += 32) {
    __syncthreads();
    // ---- cooperative staging: K tile async -> LDS, V tile transposed
    {
      const f16* ksrc = kh + ((size_t)bh * N_ + k0 + krow) * HD_ + d0;
      const f16* vsrc = vh + ((size_t)bh * N_ + k0 + krow) * HD_ + d0;
      f16*       kdst = &Ks[krow][d0];
#if __has_builtin(__builtin_amdgcn_global_load_async_to_lds_b128)
      __builtin_amdgcn_global_load_async_to_lds_b128(
          (AS1 i4v*)ksrc, (AS3 i4v*)kdst, 0, 0);
#else
      *(uint4*)kdst = *(const uint4*)ksrc;
#endif
      F16x8 vv;
      vv.u = *(const uint4*)vsrc;
#pragma unroll
      for (int j = 0; j < 8; ++j) Vt[d0 + j][krow] = vv.h[j];
      if (k0 + 32 < N_) {                       // prefetch next tiles
        __builtin_prefetch(ksrc + 32 * HD_, 0, 0);
        __builtin_prefetch(vsrc + 32 * HD_, 0, 0);
      }
#if __has_builtin(__builtin_amdgcn_s_wait_asynccnt)
      __builtin_amdgcn_s_wait_asynccnt(0);
#else
      asm volatile("s_wait_asynccnt 0" ::: "memory");
#endif
    }
    __syncthreads();

    // ---- S = Q * K^T  (two 16x16 column tiles, K-dim = HD = 64)
    v8f s0 = {}, s1 = {};
#pragma unroll
    for (int w = 0; w < 2; ++w) {
      F16Frag b0, b1;
      const f16* kp0 = &Ks[ln][w * 32 + hi * 8];
      const f16* kp1 = &Ks[16 + ln][w * 32 + hi * 8];
      b0.u[0] = *(const uint4*)kp0;  b0.u[1] = *(const uint4*)(kp0 + 16);
      b1.u[0] = *(const uint4*)kp1;  b1.u[1] = *(const uint4*)(kp1 + 16);
      v16h a = w ? qf1.v : qf0.v;
      s0 = wmma_f16(a, b0.v, s0);
      s1 = wmma_f16(a, b1.v, s1);
    }

    // ---- streaming softmax (rows live across 16-lane halves)
#pragma unroll
    for (int r = 0; r < 8; ++r) {
      int   q  = q0 + r + hi * 8;
      float v0 = s0[r] * scale + pb[(size_t)q * N_ + k0 + ln];
      float v1 = s1[r] * scale + pb[(size_t)q * N_ + k0 + 16 + ln];
      float tm = fmaxf(v0, v1);
#pragma unroll
      for (int m = 1; m < 16; m <<= 1) tm = fmaxf(tm, __shfl_xor(tm, m, 32));
      float newm = fmaxf(mrow[r], tm);
      float corr = __expf(mrow[r] - newm);
      float p0   = __expf(v0 - newm);
      float p1   = __expf(v1 - newm);
      float rs   = p0 + p1;
#pragma unroll
      for (int m = 1; m < 16; m <<= 1) rs += __shfl_xor(rs, m, 32);
      lrow[r] = lrow[r] * corr + rs;
      mrow[r] = newm;
      acc[0][r] *= corr; acc[1][r] *= corr;
      acc[2][r] *= corr; acc[3][r] *= corr;
      Ps[wave][r + hi * 8][ln]      = (f16)p0;   // C-layout -> LDS
      Ps[wave][r + hi * 8][16 + ln] = (f16)p1;
    }
    asm volatile("s_wait_dscnt 0" ::: "memory"); // P stores visible to reads

    // ---- P back as A-fragment (16x32), then acc += P * V
    F16Frag pf;
    const f16* pp = &Ps[wave][ln][0];
    pf.u[0] = *(const uint4*)(pp + sA);
    pf.u[1] = *(const uint4*)(pp + 16 + sA);
#pragma unroll
    for (int t = 0; t < 4; ++t) {
      F16Frag bv_;
      const f16* vp = &Vt[t * 16 + ln][hi * 8];
      bv_.u[0] = *(const uint4*)vp;
      bv_.u[1] = *(const uint4*)(vp + 16);
      acc[t] = wmma_f16(pf.v, bv_.v, acc[t]);
    }
  }

  // ---- epilogue: normalize and write f16 [B, N, D]
#pragma unroll
  for (int t = 0; t < 4; ++t) {
#pragma unroll
    for (int r = 0; r < 8; ++r) {
      int q = q0 + r + hi * 8;
      int d = t * 16 + ln;
      oh[((size_t)b * N_ + q) * D_ + h * HD_ + d] = (f16)(acc[t][r] / lrow[r]);
    }
  }
}

// ----------------------------------------------- output projection (f32 out)
// Same 16x64-strip WMMA GEMM, f32 result + bias straight to d_out.
__global__ __launch_bounds__(256) void gemm_out(
    const f16* __restrict__ ah, const f16* __restrict__ woh,
    const float* __restrict__ bo, float* __restrict__ out) {
  const int lane = threadIdx.x & 31;
  const int wave = threadIdx.x >> 5;
  const int gw   = blockIdx.x * 8 + wave;
  const int mt   = gw >> 4;
  const int ns   = gw & 15;
  const int ln   = lane & 15;
  const int hi   = lane >> 4;
  const int sA   = hi ? 8 : 0;

  const int arow = mt * 16 + ln;

  v8f acc[4] = {};
  const f16* ap  = ah  + (size_t)arow * D_ + sA;
  const f16* bp0 = woh + (size_t)(ns * 64 + ln) * D_ + hi * 8;
  for (int kk = 0; kk < D_; kk += 32) {
    F16Frag a, b[4];
    a.u[0] = *(const uint4*)(ap + kk);
    a.u[1] = *(const uint4*)(ap + kk + 16);
#pragma unroll
    for (int t = 0; t < 4; ++t) {
      const f16* bp = bp0 + (size_t)t * 16 * D_ + kk;
      b[t].u[0] = *(const uint4*)(bp);
      b[t].u[1] = *(const uint4*)(bp + 16);
    }
#pragma unroll
    for (int t = 0; t < 4; ++t) acc[t] = wmma_f16(a.v, b[t].v, acc[t]);
  }

#pragma unroll
  for (int t = 0; t < 4; ++t) {
    const int   col = ns * 64 + t * 16 + ln;
    const float bb  = bo[col];
#pragma unroll
    for (int r = 0; r < 8; ++r) {
      int m = mt * 16 + r + hi * 8;
      out[(size_t)m * D_ + col] = acc[t][r] + bb;
    }
  }
}

// ---------------------------------------------------------------------- host
extern "C" void kernel_launch(void* const* d_in, const int* in_sizes, int n_in,
                              void* d_out, int out_size, void* d_ws,
                              size_t ws_size, hipStream_t stream) {
  const float* x  = (const float*)d_in[0];
  const float* pb = (const float*)d_in[1];
  const float* Wq = (const float*)d_in[2];
  const float* bq = (const float*)d_in[3];
  const float* Wk = (const float*)d_in[4];
  const float* bk = (const float*)d_in[5];
  const float* Wv = (const float*)d_in[6];
  const float* bv = (const float*)d_in[7];
  const float* Wo = (const float*)d_in[8];
  const float* bo = (const float*)d_in[9];
  float* out = (float*)d_out;

  char* ws  = (char*)d_ws;
  f16* xh   = (f16*)(ws + 0);                    //  8 MB : x  f16
  f16* wqh  = (f16*)(ws + (8u  << 20));          //  2 MB
  f16* wkh  = (f16*)(ws + (10u << 20));          //  2 MB
  f16* wvh  = (f16*)(ws + (12u << 20));          //  2 MB
  f16* woh  = (f16*)(ws + (14u << 20));          //  2 MB
  f16* qh   = (f16*)(ws + (16u << 20));          //  8 MB : [B,H,N,HD]
  f16* kh   = (f16*)(ws + (24u << 20));          //  8 MB
  f16* vh   = (f16*)(ws + (32u << 20));          //  8 MB
  f16* oh   = (f16*)(ws + (40u << 20));          //  8 MB : attn out [B,N,D]

  const int nX = B_ * N_ * D_;                   // 4194304
  const int nW = D_ * D_;                        // 1048576
  cvt_f32_f16<<<(nX + 255) / 256, 256, 0, stream>>>(x,  xh,  nX);
  cvt_f32_f16<<<(nW + 255) / 256, 256, 0, stream>>>(Wq, wqh, nW);
  cvt_f32_f16<<<(nW + 255) / 256, 256, 0, stream>>>(Wk, wkh, nW);
  cvt_f32_f16<<<(nW + 255) / 256, 256, 0, stream>>>(Wv, wvh, nW);
  cvt_f32_f16<<<(nW + 255) / 256, 256, 0, stream>>>(Wo, woh, nW);

  // 256 M-tiles x 16 N-strips = 4096 waves, 8 waves/block
  gemm_qkv<<<dim3(512, 1, 3), 256, 0, stream>>>(xh, wqh, wkh, wvh, bq, bk, bv,
                                                qh, kh, vh);

  // 16 q-blocks of 128 rows per (b,h); 32 (b,h) pairs
  attn<<<dim3(16, B_ * H_), 256, 0, stream>>>(qh, kh, vh, pb, oh);

  gemm_out<<<dim3(512), 256, 0, stream>>>(oh, woh, bo, out);
}
